// GGNNModel_13443247636582
// MI455X (gfx1250) — matrix operand order repaired and checked by reading, not compiled, not run on hardware
//
#include <hip/hip_runtime.h>
#include <cstdint>
#include <cstddef>

#define NN      200000
#define DD      128
#define NET     3
#define ME      500000
#define ESL     6               // edge slots (3 types x 2 directions)
#define MSGK    (ESL * DD)      // 768
#define KST_MSG (MSGK / 32)     // 24 k-steps for msgs GEMM
#define NT_MSG  (DD / 16)       // 8 n-tiles (128 cols)
#define GRUE    (3 * DD)        // 384
#define NT_GRU  (GRUE / 16)     // 24 n-tiles
#define KST_GRU (DD / 32)       // 4 k-steps

typedef __bf16 bf16_t;
typedef bf16_t v16bf __attribute__((ext_vector_type(16)));
typedef bf16_t v8bf  __attribute__((ext_vector_type(8)));
typedef float  v8f   __attribute__((ext_vector_type(8)));
typedef unsigned int u32x4 __attribute__((ext_vector_type(4)));
typedef int i32x8 __attribute__((ext_vector_type(8)));
typedef int i32x4 __attribute__((ext_vector_type(4)));

// --- CDNA5 async data-movement feature detection (device pass only) ---------
// Preference: TDM 5-arg builtin (ROCm 7.2 / clang-22 shape; the 6-arg
// toolchain is identified by it shipping the gfx1250 TDM header), then the
// upstream async-to-LDS builtins, then a manual vectorized copy.
#if defined(__HIP_DEVICE_COMPILE__) && defined(__gfx1250__)
#if !__has_include(<hip/amd_detail/amd_gfx1250_TDM.h>)
#if __has_builtin(__builtin_amdgcn_tensor_load_to_lds) && __has_builtin(__builtin_amdgcn_s_wait_tensorcnt)
#define USE_TDM5 1
#endif
#endif
#if !defined(USE_TDM5)
#if __has_builtin(__builtin_amdgcn_global_load_async_to_lds_b128) && __has_builtin(__builtin_amdgcn_s_wait_asynccnt)
#define USE_ASYNC_LDS 1
#endif
#endif
#endif

#ifdef USE_ASYNC_LDS
// Exact builtin parameter types per the hipcc diagnostic:
//   arg0: v4i32 addrspace(1)*  (global source, 16B granule)
//   arg1: v4i32 addrspace(3)*  (LDS destination)
typedef int v4i_t __attribute__((ext_vector_type(4)));
typedef __attribute__((address_space(1))) v4i_t gv4i_t;
typedef __attribute__((address_space(3))) v4i_t lv4i_t;
#endif

// A-fragment loader: contiguous k order (k = 16*g + s), matching the packed B frags.
// lane<16: M=lane, group g=0 ; lane>=16: M=lane-16, g=1.
// elements 0..15 = A[M][kbase + g*16 .. +16)
static __device__ inline v16bf load_a_frag(const bf16_t* rowPtrAtKbase, int g) {
  union { v16bf v; v8bf h[2]; } u;
  u.h[0] = *(const v8bf*)(rowPtrAtKbase + g * 16);
  u.h[1] = *(const v8bf*)(rowPtrAtKbase + g * 16 + 8);
  return u.v;
}

static __device__ inline float sigmoidf_(float x) {
  return 1.0f / (1.0f + __expf(-x));
}

#ifdef USE_TDM5
// Issue a 1-D TDM load: nElem bf16 elements from gaddr -> LDS byte offset.
static __device__ inline void tdm_load_1d(uint64_t gaddr, uint32_t lds_off,
                                          uint32_t nElem) {
  u32x4 g0;
  g0[0] = 1u;                                   // count=1, user mode
  g0[1] = lds_off;                              // lds_addr (bytes)
  g0[2] = (uint32_t)gaddr;                      // global_addr[31:0]
  g0[3] = ((uint32_t)(gaddr >> 32) & 0x01FFFFFFu) | (2u << 30);  // [56:32] | type=2
  i32x8 g1;
  g1[0] = 0x00010000;                           // wg_mask=0, data_size=1 (2B)
  g1[1] = (int)((nElem & 0xFFFFu) << 16);       // tensor_dim0[15:0]
  g1[2] = (int)(((nElem >> 16) & 0xFFFFu) | (1u << 16));  // dim0 hi | tensor_dim1=1
  g1[3] = (int)(nElem << 16);                   // tile_dim0 = nElem
  g1[4] = 1;                                    // tile_dim1 = 1, tile_dim2 = 0
  g1[5] = (int)nElem;                           // tensor_dim0_stride[31:0]
  g1[6] = (int)((nElem & 0xFFFFu) << 16);       // stride0 hi=0 | stride1[15:0]
  g1[7] = (int)(nElem >> 16);                   // stride1 hi
  i32x4 z = {0, 0, 0, 0};
  __builtin_amdgcn_tensor_load_to_lds(g0, g1, z, z, 0);
}
#endif

// ---------------------------------------------------------------- zero
__global__ __launch_bounds__(256) void k_zero_f4(float4* __restrict__ p, long long n4) {
  long long i = (long long)blockIdx.x * 256 + threadIdx.x;
  if (i < n4) p[i] = make_float4(0.f, 0.f, 0.f, 0.f);
}

// ---------------------------------------------------------------- init h
__global__ __launch_bounds__(256) void k_init_h(const float* __restrict__ ns,
                                                float* __restrict__ hf,
                                                bf16_t* __restrict__ hb, long long n) {
  long long i = (long long)blockIdx.x * 256 + threadIdx.x;
  if (i < n) { float v = ns[i]; hf[i] = v; hb[i] = (bf16_t)v; }
}

// ------------------------------------------- pack msgs-GEMM B fragments
// Wbig[m][dout] = W_t[(m/128)*128 + dout][m%128]  (block-transposed W_t)
// frag layout: [kstep(24)][ntile(8)][lane(32)][j(16)]
__global__ __launch_bounds__(256) void k_prep_wmsg(const float* __restrict__ W_t,
                                                   bf16_t* __restrict__ frag) {
  int i = blockIdx.x * 256 + threadIdx.x;
  if (i >= KST_MSG * NT_MSG * 32 * 16) return;
  int j = i & 15;
  int lane = (i >> 4) & 31;
  int ntile = (i >> 9) & 7;
  int kstep = i >> 12;
  int g = lane >> 4;
  int n = ntile * 16 + (lane & 15);
  int kg = kstep * 32 + g * 16 + j;       // global K in [0,768)
  int slot = kg >> 7;
  int k = kg & 127;
  frag[i] = (bf16_t)W_t[(size_t)(slot * DD + n) * DD + k];
}

// ------------------------------------------- pack GRU weight B fragments
// B[k][e] = w[e][k] ; frag layout [kstep(4)][ntile(24)][lane(32)][j(16)]
__global__ __launch_bounds__(256) void k_prep_wgru(const float* __restrict__ w,
                                                   bf16_t* __restrict__ frag) {
  int i = blockIdx.x * 256 + threadIdx.x;
  if (i >= KST_GRU * NT_GRU * 32 * 16) return;
  int j = i & 15;
  int lane = (i >> 4) & 31;
  int nt_ks = i >> 9;                     // kstep*24 + ntile
  int ntile = nt_ks % NT_GRU;
  int kstep = nt_ks / NT_GRU;
  int g = lane >> 4;
  int e = ntile * 16 + (lane & 15);
  int k = kstep * 32 + g * 16 + j;
  frag[i] = (bf16_t)w[(size_t)e * DD + k];
}

// ---------------------------------------------------------------- scatter
__global__ __launch_bounds__(256) void k_scatter(const float* __restrict__ h,
                                                 const long long* __restrict__ edges,
                                                 float* __restrict__ Agg,
                                                 float* __restrict__ cnt6) {
  int wid = blockIdx.x * 8 + (threadIdx.x >> 5);
  int lane = threadIdx.x & 31;
  if (wid >= NET * ME) return;
  int type = wid / ME;
  int e = wid - type * ME;
  const long long* ep = edges + ((long long)type * ME + e) * 2;
  int s = (int)ep[0];
  int t = (int)ep[1];
  float4 hs = *(const float4*)(h + (size_t)s * DD + lane * 4);
  float4 ht = *(const float4*)(h + (size_t)t * DD + lane * 4);
  float* at = Agg + (size_t)t * MSGK + type * DD + lane * 4;
  float* as = Agg + (size_t)s * MSGK + (type + 3) * DD + lane * 4;
  atomicAdd(at + 0, hs.x); atomicAdd(at + 1, hs.y);
  atomicAdd(at + 2, hs.z); atomicAdd(at + 3, hs.w);
  atomicAdd(as + 0, ht.x); atomicAdd(as + 1, ht.y);
  atomicAdd(as + 2, ht.z); atomicAdd(as + 3, ht.w);
  if (lane == 0) {
    atomicAdd(cnt6 + (size_t)t * ESL + type, 1.0f);
    atomicAdd(cnt6 + (size_t)s * ESL + type + 3, 1.0f);
  }
}

// ------------------------------------------------------------- msgs GEMM
// msgs = Agg[N,768] @ Wbig[768,128], + count-weighted bias, / degree -> bf16
// Double-buffered A tile in LDS: one barrier per k-step.
__global__ __launch_bounds__(256) void k_gemm_msgs(const float* __restrict__ Agg,
                                                   const float* __restrict__ cnt6,
                                                   const bf16_t* __restrict__ Wfrag,
                                                   const float* __restrict__ b_t,
                                                   bf16_t* __restrict__ msgs_bf) {
  __shared__ bf16_t Atile[2][64][32];
  int tid = threadIdx.x;
  int lane = tid & 31;
  int wave = tid >> 5;
  int waveM = wave >> 1;      // 0..3
  int ng = wave & 1;          // 0..1 (covers n-tiles ng*4 .. ng*4+3)
  int rowBase = blockIdx.x * 64;
  int g = lane >> 4;

  // staging assignment: 4 threads per row, 8 consecutive k each
  int sr = tid >> 2;          // 0..63
  int sk = (tid & 3) * 8;     // 0,8,16,24
  const float* srcRow = &Agg[(size_t)(rowBase + sr) * MSGK + sk];

  auto stage = [&](int kstep, int buf) {
    const float4* src = (const float4*)(srcRow + kstep * 32);
    float4 a0 = src[0], a1 = src[1];
    v8bf pk;
    pk[0] = (bf16_t)a0.x; pk[1] = (bf16_t)a0.y;
    pk[2] = (bf16_t)a0.z; pk[3] = (bf16_t)a0.w;
    pk[4] = (bf16_t)a1.x; pk[5] = (bf16_t)a1.y;
    pk[6] = (bf16_t)a1.z; pk[7] = (bf16_t)a1.w;
    *(v8bf*)&Atile[buf][sr][sk] = pk;
  };

  v8f acc[4] = {};

  stage(0, 0);
  __syncthreads();

  for (int kstep = 0; kstep < KST_MSG; ++kstep) {
    int cur = kstep & 1;
    if (kstep + 1 < KST_MSG) {
      stage(kstep + 1, cur ^ 1);
      if (kstep + 2 < KST_MSG)
        __builtin_prefetch(srcRow + (kstep + 2) * 32, 0, 0);
    }
    v16bf a = load_a_frag(&Atile[cur][waveM * 16 + (lane & 15)][0], g);
#pragma unroll
    for (int nt = 0; nt < 4; ++nt) {
      int ntile = ng * 4 + nt;
      v16bf b = *(const v16bf*)(Wfrag + (((size_t)kstep * NT_MSG + ntile) * 32 + lane) * 16);
      acc[nt] = __builtin_amdgcn_wmma_f32_16x16x32_bf16(
          false, a, false, b, (short)0, acc[nt], false, false);
    }
    __syncthreads();
  }

  // epilogue: bias from per-slot counts, divide by degree, store bf16
  int hi = lane >> 4;
  int nn = lane & 15;
  for (int r = 0; r < 8; ++r) {
    int row = rowBase + waveM * 16 + r + 8 * hi;
    const float* cp = cnt6 + (size_t)row * ESL;
    float cs[ESL];
    float tot = 0.f;
    for (int s = 0; s < ESL; ++s) { cs[s] = cp[s]; tot += cs[s]; }
    float div = (tot == 0.f) ? 1.f : tot;
#pragma unroll
    for (int nt = 0; nt < 4; ++nt) {
      int col = (ng * 4 + nt) * 16 + nn;
      float bias = 0.f;
      for (int s = 0; s < ESL; ++s) bias += cs[s] * b_t[s * DD + col];
      float val = (acc[nt][r] + bias) / div;
      msgs_bf[(size_t)row * DD + col] = (bf16_t)val;
    }
  }
}

// ------------------------------------------------------------- fused GRU
// Dynamic LDS layout (so TDM D#.lds_addr offsets are known):
//   [0,4096)      mT  16x128 bf16
//   [4096,8192)   hT  16x128 bf16
//   [8192,32768)  gI  16x384 f32
//   [32768,57344) gH  16x384 f32
#define GRU_LDS_BYTES 57344
__global__ __launch_bounds__(256) void k_gru(const bf16_t* __restrict__ m_bf,
                                             bf16_t* __restrict__ h_bf,
                                             float* __restrict__ h_f32,
                                             const bf16_t* __restrict__ fih,
                                             const bf16_t* __restrict__ fhh,
                                             const float* __restrict__ b_ih,
                                             const float* __restrict__ b_hh,
                                             float* __restrict__ d_out,
                                             int writeOut) {
  extern __shared__ __align__(16) char smem[];
  bf16_t* mT = (bf16_t*)smem;                       // [16][128]
  bf16_t* hT = (bf16_t*)(smem + 4096);              // [16][128]
  float*  gI = (float*)(smem + 8192);               // [16][384]
  float*  gH = (float*)(smem + 8192 + 24576);       // [16][384]

  int tid = threadIdx.x;
  int lane = tid & 31;
  int wave = tid >> 5;
  int row0 = blockIdx.x * 16;
  int g = lane >> 4;

#if defined(USE_TDM5)
  // Tensor Data Mover: one wave issues both 4KB tile loads asynchronously.
  if (tid == 0) {
    tdm_load_1d((uint64_t)(uintptr_t)(m_bf + (size_t)row0 * DD), 0u, 2048u);
    tdm_load_1d((uint64_t)(uintptr_t)(h_bf + (size_t)row0 * DD), 4096u, 2048u);
  }
  __builtin_amdgcn_s_wait_tensorcnt(0);
  __syncthreads();
#elif defined(USE_ASYNC_LDS)
  // GLOBAL_LOAD_ASYNC_TO_LDS_B128: one 16B chunk per thread per matrix.
  {
    int r = tid >> 4;            // 16 rows, 16 threads/row
    int c = (tid & 15) * 8;      // 8 bf16 each
    __builtin_amdgcn_global_load_async_to_lds_b128(
        (gv4i_t*)(m_bf + (size_t)(row0 + r) * DD + c),
        (lv4i_t*)&mT[r * DD + c], 0, 0);
    __builtin_amdgcn_global_load_async_to_lds_b128(
        (gv4i_t*)(h_bf + (size_t)(row0 + r) * DD + c),
        (lv4i_t*)&hT[r * DD + c], 0, 0);
    __builtin_amdgcn_s_wait_asynccnt(0);
  }
  __syncthreads();
#else
  // manual staging: one b128 load + b128 LDS store per thread per matrix
  {
    int r = tid >> 4;            // 16 rows, 16 threads/row
    int c = (tid & 15) * 8;      // 8 bf16 each
    *(v8bf*)&mT[r * DD + c] = *(const v8bf*)&m_bf[(size_t)(row0 + r) * DD + c];
    *(v8bf*)&hT[r * DD + c] = *(const v8bf*)&h_bf[(size_t)(row0 + r) * DD + c];
  }
  __syncthreads();
#endif

  // WMMA tiles: 8 waves x 6 tiles = 48 (24 gi + 24 gh)
#pragma unroll
  for (int q = 0; q < 6; ++q) {
    int tt = wave * 6 + q;            // 0..47
    int mat = tt / NT_GRU;            // 0 = gi, 1 = gh
    int ntile = tt % NT_GRU;
    const bf16_t* At = mat ? hT : mT;
    const bf16_t* frag = mat ? fhh : fih;
    v8f acc = {};
#pragma unroll
    for (int kstep = 0; kstep < KST_GRU; ++kstep) {
      v16bf a = load_a_frag(&At[(lane & 15) * DD + kstep * 32], g);
      v16bf b = *(const v16bf*)(frag + (((size_t)kstep * NT_GRU + ntile) * 32 + lane) * 16);
      acc = __builtin_amdgcn_wmma_f32_16x16x32_bf16(
          false, a, false, b, (short)0, acc, false, false);
    }
    float* Gt = mat ? gH : gI;
    int col = ntile * 16 + (lane & 15);
    int rbase = 8 * (lane >> 4);
#pragma unroll
    for (int r = 0; r < 8; ++r) Gt[(rbase + r) * GRUE + col] = acc[r];
  }
  __syncthreads();

  // elementwise GRU gates
  for (int j = 0; j < 8; ++j) {
    int idx = j * 256 + tid;
    int r = idx >> 7, c = idx & 127;
    int row = row0 + r;
    float ir = gI[r * GRUE + c] + b_ih[c];
    float iz = gI[r * GRUE + c + DD] + b_ih[c + DD];
    float in = gI[r * GRUE + c + 2 * DD] + b_ih[c + 2 * DD];
    float hr = gH[r * GRUE + c] + b_hh[c];
    float hz = gH[r * GRUE + c + DD] + b_hh[c + DD];
    float hn = gH[r * GRUE + c + 2 * DD] + b_hh[c + 2 * DD];
    float rg = sigmoidf_(ir + hr);
    float z  = sigmoidf_(iz + hz);
    float n  = tanhf(in + rg * hn);
    float hprev = h_f32[(size_t)row * DD + c];
    float hnew = (1.f - z) * n + z * hprev;
    h_f32[(size_t)row * DD + c] = hnew;
    h_bf[(size_t)row * DD + c] = (bf16_t)hnew;
    if (writeOut) d_out[(size_t)row * DD + c] = hnew;
  }
}

// ---------------------------------------------------------------- launch
extern "C" void kernel_launch(void* const* d_in, const int* in_sizes, int n_in,
                              void* d_out, int out_size, void* d_ws, size_t ws_size,
                              hipStream_t stream) {
  (void)in_sizes; (void)n_in; (void)out_size; (void)ws_size;
  const float*     node_states = (const float*)d_in[0];
  const long long* edges       = (const long long*)d_in[1];
  const float*     W_t         = (const float*)d_in[2];
  const float*     b_t         = (const float*)d_in[3];
  const float*     w_ih        = (const float*)d_in[4];
  const float*     w_hh        = (const float*)d_in[5];
  const float*     b_ih        = (const float*)d_in[6];
  const float*     b_hh        = (const float*)d_in[7];
  float* out = (float*)d_out;

  char* ws = (char*)d_ws;
  size_t off = 0;
  auto alloc = [&](size_t bytes) -> void* {
    void* p = ws + off;
    off = (off + bytes + 255) & ~(size_t)255;
    return p;
  };
  float*  Agg     = (float*)alloc((size_t)NN * MSGK * sizeof(float));   // 614 MB
  float*  cnt6    = (float*)alloc((size_t)NN * ESL * sizeof(float));
  bf16_t* msgs_bf = (bf16_t*)alloc((size_t)NN * DD * sizeof(bf16_t));
  float*  h_f32   = (float*)alloc((size_t)NN * DD * sizeof(float));
  bf16_t* h_bf    = (bf16_t*)alloc((size_t)NN * DD * sizeof(bf16_t));
  bf16_t* Wfrag   = (bf16_t*)alloc((size_t)MSGK * DD * sizeof(bf16_t));
  bf16_t* fih     = (bf16_t*)alloc((size_t)GRUE * DD * sizeof(bf16_t));
  bf16_t* fhh     = (bf16_t*)alloc((size_t)GRUE * DD * sizeof(bf16_t));

  // weight packing
  k_prep_wmsg<<<(KST_MSG * NT_MSG * 32 * 16 + 255) / 256, 256, 0, stream>>>(W_t, Wfrag);
  k_prep_wgru<<<(KST_GRU * NT_GRU * 32 * 16 + 255) / 256, 256, 0, stream>>>(w_ih, fih);
  k_prep_wgru<<<(KST_GRU * NT_GRU * 32 * 16 + 255) / 256, 256, 0, stream>>>(w_hh, fhh);

  // h0
  long long nh = (long long)NN * DD;
  k_init_h<<<(int)((nh + 255) / 256), 256, 0, stream>>>(node_states, h_f32, h_bf, nh);

  for (int t = 0; t < 2; ++t) {
    long long nAgg4 = (long long)NN * MSGK / 4;
    long long nCnt4 = (long long)NN * ESL / 4;
    k_zero_f4<<<(int)((nAgg4 + 255) / 256), 256, 0, stream>>>((float4*)Agg, nAgg4);
    k_zero_f4<<<(int)((nCnt4 + 255) / 256), 256, 0, stream>>>((float4*)cnt6, nCnt4);

    int nWaves = NET * ME;                       // 1.5M edge-waves
    k_scatter<<<(nWaves + 7) / 8, 256, 0, stream>>>(h_f32, edges, Agg, cnt6);

    k_gemm_msgs<<<NN / 64, 256, 0, stream>>>(Agg, cnt6, Wfrag, b_t, msgs_bf);

    k_gru<<<NN / 16, 256, GRU_LDS_BYTES, stream>>>(msgs_bf, h_bf, h_f32, fih, fhh,
                                                   b_ih, b_hh, out, (t == 1) ? 1 : 0);
  }
}
